// nOrderModule_88905823027220
// MI455X (gfx1250) — matrix operand order repaired and checked by reading, not compiled, not run on hardware
//
#include <hip/hip_runtime.h>

// C[512,128] = X[512, 1M] @ W[1M, 128]   (einsum 'bij,ijk->bk' flattened)
#define BATCH   512
#define KOUT    128
#define KRED    (1024*1024)
#define MWG     64                  // M rows per workgroup (4 wmma tiles)
#define NMBLK   (BATCH / MWG)       // 8
#define KCHUNK  8192
#define NKCHUNK (KRED / KCHUNK)     // 128
#define KSTEP   64

typedef __attribute__((ext_vector_type(16))) __bf16        v16bf;
typedef __attribute__((ext_vector_type(8)))  float         v8f;
typedef __attribute__((ext_vector_type(4))) unsigned int   u32x4;
typedef __attribute__((ext_vector_type(2))) unsigned int   u32x2;
typedef __attribute__((ext_vector_type(4)))  float         f32x4;
typedef __attribute__((ext_vector_type(2)))  float         f32x2;
typedef __attribute__((ext_vector_type(2)))  __bf16        bf16x2;

// packed f32->bf16 pair -> v_cvt_pk_bf16_f32 (confirmed in disasm)
__device__ __forceinline__ unsigned int pack2(float lo, float hi) {
    f32x2 v; v.x = lo; v.y = hi;
    bf16x2 b = __builtin_convertvector(v, bf16x2);
    return __builtin_bit_cast(unsigned int, b);
}

__device__ __forceinline__ f32x4 ld_nt(const float* p) {
    return __builtin_nontemporal_load((const f32x4*)p);
}

__device__ __forceinline__ void sched_ds_then_wmma() {
#if __has_builtin(__builtin_amdgcn_sched_group_barrier)
    // pin schedule: 10 DS-reads (all fragments), then 4 WMMAs back-to-back
    __builtin_amdgcn_sched_group_barrier(0x100, 10, 0);  // DS read
    __builtin_amdgcn_sched_group_barrier(0x008, 4, 0);   // MFMA/WMMA
#endif
}

union Frag { u32x4 q[2]; v16bf v; };

__global__ __launch_bounds__(256) void gemm_splitk_bf16(
        const float* __restrict__ A,     // [512][1M]
        const float* __restrict__ Bm,    // [1M][128]
        float* __restrict__ out)         // [512][128]
{
    // padded rows: 64 bf16 + 8 pad -> 144B stride (16B aligned, conflict-light)
    __shared__ alignas(16) unsigned short sA[MWG][72];
    __shared__ alignas(16) unsigned short sB[KOUT][72];

    const int t    = threadIdx.x;        // 0..255
    const int lane = t & 31;
    const int wave = t >> 5;             // 0..7 -> n-tile
    const int mbWG = blockIdx.x;         // 0..7
    const long long kBase = (long long)blockIdx.y * KCHUNK;

    // A loader: each thread loads one row, 16 contiguous k (4 x b128)
    const int la_m = t >> 2;             // 0..63
    const int la_k = (t & 3) * 16;       // 0,16,32,48
    // B loader: each thread loads 4 consecutive k-rows x 8 contiguous n
    const int lb_k = (t >> 4) * 4;       // 0..60
    const int lb_n = (t & 15) * 8;       // 0..120

    const float* aBase = A  + (long long)(mbWG * MWG + la_m) * KRED + kBase + la_k;
    const float* bBase = Bm + kBase * KOUT;

    const int lrow = lane & 15;
    const int lhi  = lane >> 4;

    v8f acc[4];
#pragma unroll
    for (int mt = 0; mt < 4; ++mt)
        acc[mt] = (v8f){0.f, 0.f, 0.f, 0.f, 0.f, 0.f, 0.f, 0.f};

    for (int kk = 0; kk < KCHUNK; kk += KSTEP) {
        // ---------- global loads into registers (coalesced b128) ----------
        f32x4 a0 = ld_nt(aBase + kk);
        f32x4 a1 = ld_nt(aBase + kk + 4);
        f32x4 a2 = ld_nt(aBase + kk + 8);
        f32x4 a3 = ld_nt(aBase + kk + 12);

        const float* bp = bBase + (long long)(kk + lb_k) * KOUT + lb_n;
        f32x4 b0a = *(const f32x4*)(bp);
        f32x4 b0b = *(const f32x4*)(bp + 4);
        f32x4 b1a = *(const f32x4*)(bp + KOUT);
        f32x4 b1b = *(const f32x4*)(bp + KOUT + 4);
        f32x4 b2a = *(const f32x4*)(bp + 2 * KOUT);
        f32x4 b2b = *(const f32x4*)(bp + 2 * KOUT + 4);
        f32x4 b3a = *(const f32x4*)(bp + 3 * KOUT);
        f32x4 b3b = *(const f32x4*)(bp + 3 * KOUT + 4);

        if (kk + KSTEP < KCHUNK) {
            __builtin_prefetch(bp + (long long)KSTEP * KOUT, 0, 0);
            __builtin_prefetch(aBase + kk + KSTEP, 0, 0);
        }

        __syncthreads();   // previous iteration's fragment reads done

        // ---------- stage A tile as bf16 (row-major [m][k]): 2 x ds_store_b128 ----------
        {
            u32x4 qa0 = { pack2(a0.x, a0.y), pack2(a0.z, a0.w),
                          pack2(a1.x, a1.y), pack2(a1.z, a1.w) };
            u32x4 qa1 = { pack2(a2.x, a2.y), pack2(a2.z, a2.w),
                          pack2(a3.x, a3.y), pack2(a3.z, a3.w) };
            *((u32x4*)&sA[la_m][la_k])     = qa0;
            *((u32x4*)&sA[la_m][la_k + 8]) = qa1;
        }
        // ---------- stage B transposed as bf16 ([n][k]): 8 x ds_store_b64 ----------
        {
            float r0[8] = {b0a.x, b0a.y, b0a.z, b0a.w, b0b.x, b0b.y, b0b.z, b0b.w};
            float r1[8] = {b1a.x, b1a.y, b1a.z, b1a.w, b1b.x, b1b.y, b1b.z, b1b.w};
            float r2[8] = {b2a.x, b2a.y, b2a.z, b2a.w, b2b.x, b2b.y, b2b.z, b2b.w};
            float r3[8] = {b3a.x, b3a.y, b3a.z, b3a.w, b3b.x, b3b.y, b3b.z, b3b.w};
#pragma unroll
            for (int i = 0; i < 8; ++i) {
                u32x2 v;
                v.x = pack2(r0[i], r1[i]);   // k = lb_k, lb_k+1
                v.y = pack2(r2[i], r3[i]);   // k = lb_k+2, lb_k+3
                *((u32x2*)&sB[lb_n + i][lb_k]) = v;
            }
        }

        __syncthreads();

        // ---------- per k-half: ALL fragment loads, then 4 back-to-back WMMAs ----------
#pragma unroll
        for (int h = 0; h < 2; ++h) {
            const int eb = h * 32 + lhi * 8;        // B: K = lhi*8+{0..7}, +16
            const int ea = h * 32 + lhi * 8;        // A: K = lhi*8+{0..7}, +16
            const int n  = wave * 16 + lrow;

            Frag fb, fa0, fa1, fa2, fa3;
            fb.q[0]  = *((const u32x4*)&sB[n][eb]);
            fb.q[1]  = *((const u32x4*)&sB[n][eb + 16]);
            fa0.q[0] = *((const u32x4*)&sA[lrow][ea]);
            fa0.q[1] = *((const u32x4*)&sA[lrow][ea + 16]);
            fa1.q[0] = *((const u32x4*)&sA[16 + lrow][ea]);
            fa1.q[1] = *((const u32x4*)&sA[16 + lrow][ea + 16]);
            fa2.q[0] = *((const u32x4*)&sA[32 + lrow][ea]);
            fa2.q[1] = *((const u32x4*)&sA[32 + lrow][ea + 16]);
            fa3.q[0] = *((const u32x4*)&sA[48 + lrow][ea]);
            fa3.q[1] = *((const u32x4*)&sA[48 + lrow][ea + 16]);

            acc[0] = __builtin_amdgcn_wmma_f32_16x16x32_bf16(
                         false, fa0.v, false, fb.v, (short)0, acc[0], false, false);
            acc[1] = __builtin_amdgcn_wmma_f32_16x16x32_bf16(
                         false, fa1.v, false, fb.v, (short)0, acc[1], false, false);
            acc[2] = __builtin_amdgcn_wmma_f32_16x16x32_bf16(
                         false, fa2.v, false, fb.v, (short)0, acc[2], false, false);
            acc[3] = __builtin_amdgcn_wmma_f32_16x16x32_bf16(
                         false, fa3.v, false, fb.v, (short)0, acc[3], false, false);

            sched_ds_then_wmma();
        }
    }

    // ---------- split-K combine: f32 atomics into output ----------
    {
        const int col = wave * 16 + lrow;
#pragma unroll
        for (int mt = 0; mt < 4; ++mt) {
            const int rowBase = mbWG * MWG + mt * 16 + lhi * 8;
#pragma unroll
            for (int r = 0; r < 8; ++r)
                atomicAdd(&out[(rowBase + r) * KOUT + col], acc[mt][r]);
        }
    }
}

__global__ void zero_f32(float* p, int n) {
    int i = blockIdx.x * blockDim.x + threadIdx.x;
    if (i < n) p[i] = 0.0f;
}

extern "C" void kernel_launch(void* const* d_in, const int* in_sizes, int n_in,
                              void* d_out, int out_size, void* d_ws, size_t ws_size,
                              hipStream_t stream) {
    const float* A  = (const float*)d_in[0];   // x_after_expand [512,1024,1024]
    const float* Bm = (const float*)d_in[1];   // kernel_metrix  [1024,1024,128]
    float* out = (float*)d_out;                // [512,128]

    const int n = BATCH * KOUT;
    zero_f32<<<(n + 255) / 256, 256, 0, stream>>>(out, n);
    gemm_splitk_bf16<<<dim3(NMBLK, NKCHUNK), dim3(256), 0, stream>>>(A, Bm, out);
}
